// AxialAttention1d_89223650607716
// MI455X (gfx1250) — matrix operand order repaired and checked by reading, not compiled, not run on hardware
//
#include <hip/hip_runtime.h>
#include <hip/hip_bf16.h>
#include <stdint.h>

// ---------------------------------------------------------------------------
// AxialAttention1d forward for MI455X (gfx1250, wave32, WMMA).
// L=4096, N=8, C=512, OUT=512, G=8, KS=64 -> W=64, B=512 windows, GP=64.
// All matmul-shaped work (incl. relative-position einsums) on
// v_wmma_f32_16x16x32_bf16; VALU only for softmax / gathers / BN.
// ---------------------------------------------------------------------------

typedef __bf16 bf16_t;
typedef __attribute__((ext_vector_type(16))) __bf16 v16bf;
typedef __attribute__((ext_vector_type(8)))  float  v8f;

#define EPS 1e-5f

__device__ __forceinline__ bf16_t f2bf(float f) { return (bf16_t)f; }
__device__ __forceinline__ float  bf2f(bf16_t b) { return (float)b; }

union FragU { v16bf v; uint4 q[2]; };

// A-matrix fragment (16x32 bf16) from (M,K) row-major memory.
// ISA 7.12.2: lanes 0-15 hold M=lane, K 0-7 & 16-23; lanes 16-31 K 8-15 & 24-31.
__device__ __forceinline__ v16bf load_frag_a(const bf16_t* base, int stride) {
  int lane = threadIdx.x & 31;
  int r = lane & 15, koff = (lane >> 4) * 8;
  const bf16_t* p = base + r * stride + koff;
  FragU f;
  f.q[0] = *(const uint4*)(p);        // K = koff .. koff+7
  f.q[1] = *(const uint4*)(p + 16);   // K = koff+16 .. koff+23
  return f.v;
}

// B-matrix fragment (32x16 bf16) from (N,K) row-major memory.
// lanes 0-15 = column N, K 0-7 & 16-23; lanes 16-31 K 8-15 & 24-31 (matches A).
__device__ __forceinline__ v16bf load_frag_b(const bf16_t* base, int stride) {
  int lane = threadIdx.x & 31;
  int n = lane & 15, koff = (lane >> 4) * 8;
  const bf16_t* p = base + n * stride + koff;
  FragU f;
  f.q[0] = *(const uint4*)(p);        // K = koff .. koff+7
  f.q[1] = *(const uint4*)(p + 16);   // K = koff+16 .. koff+23
  return f.v;
}

// ---------------------------------------------------------------------------
// Kernel 1: QKV GEMM  qkv[b,o,h] = sum_c qkv_w[o,c] * x[w*64+h, n, c]
// grid = (8 m-blocks == head g, 512 windows), 256 threads (8 waves).
// 128(M) x 64(N) per WG, K=512 in 16 slices of 32.  BN fused; scattered into
// WMMA-layout bf16 q/k/v workspace.
// ---------------------------------------------------------------------------
__global__ __launch_bounds__(256) void qkv_gemm_kernel(
    const float* __restrict__ x, const float* __restrict__ qkv_w,
    const float* __restrict__ qkv_g, const float* __restrict__ qkv_b,
    const float* __restrict__ qkv_m, const float* __restrict__ qkv_v,
    bf16_t* __restrict__ wq, bf16_t* __restrict__ wk, bf16_t* __restrict__ wv)
{
  __shared__ __align__(16) bf16_t sA[128 * 48];   // (m, c) stride 48 (96B)
  __shared__ __align__(16) bf16_t sB[64 * 48];    // (h, c) stride 48

  const int mblk = blockIdx.x;                    // == head group g
  const int b    = blockIdx.y;                    // window index
  const int n = b >> 6, w = b & 63;
  const int tid = threadIdx.x;
  const int wave = tid >> 5, lane = tid & 31;
  const int hl = lane >> 4;

  v8f acc[4] = {};

  for (int kk = 0; kk < 16; ++kk) {
    __syncthreads();
    {
      int row = tid >> 1, half = tid & 1;
      const float* src = qkv_w + (size_t)(mblk * 128 + row) * 512 + kk * 32 + half * 16;
      bf16_t* dst = sA + row * 48 + half * 16;
      #pragma unroll
      for (int e = 0; e < 16; ++e) dst[e] = f2bf(src[e]);
    }
    if (tid < 128) {
      int h = tid >> 1, half = tid & 1;
      const float* src = x + ((size_t)(w * 64 + h) * 8 + n) * 512 + kk * 32 + half * 16;
      bf16_t* dst = sB + h * 48 + half * 16;
      #pragma unroll
      for (int e = 0; e < 16; ++e) dst[e] = f2bf(src[e]);
    }
    if (kk + 1 < 16) {
      // gfx1250 speculative prefetch of next K-slice -> global_prefetch_b8
      const float* pa = qkv_w + (size_t)(mblk * 128 + (tid >> 1)) * 512 + (kk + 1) * 32;
      __builtin_prefetch(pa, 0, 1);
    }
    __syncthreads();

    v16bf afrag = load_frag_a(sA + wave * 16 * 48, 48);
    #pragma unroll
    for (int nt = 0; nt < 4; ++nt) {
      v16bf bfrag = load_frag_b(sB + nt * 16 * 48, 48);
      acc[nt] = __builtin_amdgcn_wmma_f32_16x16x32_bf16(
          false, afrag, false, bfrag, (short)0, acc[nt], false, false);
    }
  }

  // BN epilogue + scatter:  q^T (i,c) 64x32 ; k^T (j,c) 64x32 ; v (c,j) 64x64
  const size_t bg = (size_t)b * 8 + mblk;
  #pragma unroll
  for (int r = 0; r < 8; ++r) {
    int cc = wave * 16 + r + 8 * hl;              // channel within group [0,128)
    int o  = mblk * 128 + cc;                     // qkv channel [0,1024)
    float scale = qkv_g[o] * rsqrtf(qkv_v[o] + EPS);
    float shift = qkv_b[o] - qkv_m[o] * scale;
    #pragma unroll
    for (int nt = 0; nt < 4; ++nt) {
      int h = nt * 16 + (lane & 15);
      bf16_t bv = f2bf(acc[nt][r] * scale + shift);
      if (cc < 32)      wq[bg * 2048 + (size_t)h * 32 + cc] = bv;
      else if (cc < 64) wk[bg * 2048 + (size_t)h * 32 + (cc - 32)] = bv;
      else              wv[bg * 4096 + (size_t)(cc - 64) * 64 + h] = bv;
    }
  }
}

// ---------------------------------------------------------------------------
// Kernel 2: attention per (b,g).  grid = 4096, 256 threads (8 waves).
// LDS arena reused across phases (~92 KB/WG -> 3 WG per 320KB WGP).
// ---------------------------------------------------------------------------
__global__ __launch_bounds__(256) void attn_kernel(
    const bf16_t* __restrict__ wq, const bf16_t* __restrict__ wk,
    const bf16_t* __restrict__ wv, const float* __restrict__ relative,
    const float* __restrict__ sim_g, const float* __restrict__ sim_b,
    const float* __restrict__ sim_m, const float* __restrict__ sim_v,
    const float* __restrict__ out_g, const float* __restrict__ out_b,
    const float* __restrict__ out_m, const float* __restrict__ out_v,
    float* __restrict__ out)
{
  __shared__ __align__(16) float  sSim[64 * 65];   // scores -> probs (padded)
  __shared__ __align__(16) bf16_t sSimB[64 * 72];  // bf16 probs, A-layout rows
  __shared__ __align__(16) unsigned char arena[68096];

  // Phase qr/kr views
  bf16_t* sRelQT = (bf16_t*)(arena);               // 128x32 bf16 (d,c)  8192 B
  bf16_t* sRelKT = (bf16_t*)(arena + 8192);        // 128x32 bf16 (d,c)  8192 B
  float*  sP     = (float*)(arena + 16384);        // 64x132 f32        33792 B
  // Phase sv/sve views (after softmax; previous views dead)
  bf16_t* sRelV  = (bf16_t*)(arena);               // 64x136 bf16 (c,m) 17408 B
  bf16_t* sST    = (bf16_t*)(arena + 17408);       // 64x136 bf16 (i,m) 17408 B
  float*  sSv    = (float*)(arena + 34816);        // 64x65 f32         16640 B
  float*  sSve   = (float*)(arena + 51456);        // 64x65 f32         16640 B

  const int bg = blockIdx.x;
  const int b = bg >> 3, g = bg & 7;
  const int n = b >> 6, w = b & 63;
  const int tid = threadIdx.x, wave = tid >> 5, lane = tid & 31;
  const int hl = lane >> 4;

  const bf16_t* qg = wq + (size_t)bg * 2048;   // (i,c) 64x32
  const bf16_t* kg = wk + (size_t)bg * 2048;   // (j,c) 64x32
  const bf16_t* vg = wv + (size_t)bg * 4096;   // (c,j) 64x64

  // sim BN scalars (channels g / 8+g / 16+g of the 3G stack)
  const float sc0 = sim_g[g]      * rsqrtf(sim_v[g]      + EPS);
  const float sh0 = sim_b[g]      - sim_m[g]      * sc0;
  const float sc1 = sim_g[8 + g]  * rsqrtf(sim_v[8 + g]  + EPS);
  const float sh1 = sim_b[8 + g]  - sim_m[8 + g]  * sc1;
  const float sc2 = sim_g[16 + g] * rsqrtf(sim_v[16 + g] + EPS);
  const float sh2 = sim_b[16 + g] - sim_m[16 + g] * sc2;

  // ---- Phase 1: qk[i,j] = sum_c q[c,i]k[c,j]  (one 16x16x32 WMMA per tile)
  #pragma unroll
  for (int tt = 0; tt < 2; ++tt) {
    int t = wave * 2 + tt;
    int it = t >> 2, jt = t & 3;
    v16bf a  = load_frag_a(qg + it * 16 * 32, 32);
    v16bf bb = load_frag_b(kg + jt * 16 * 32, 32);
    v8f c = {};
    c = __builtin_amdgcn_wmma_f32_16x16x32_bf16(false, a, false, bb, (short)0, c, false, false);
    #pragma unroll
    for (int r = 0; r < 8; ++r) {
      int i = it * 16 + r + 8 * hl;
      int j = jt * 16 + (lane & 15);
      sSim[i * 65 + j] = c[r] * sc0 + sh0;
    }
  }

  // ---- Build transposed relative tables (B-layout, row d, col c), pad row 127=0
  #pragma unroll 1
  for (int ii = 0; ii < 16; ++ii) {
    int e = tid + 256 * ii;                       // 4096 entries
    int d = e >> 5, c = e & 31;
    float vq = (d < 127) ? relative[c * 127 + d]        : 0.f;
    float vk = (d < 127) ? relative[(32 + c) * 127 + d] : 0.f;
    sRelQT[d * 32 + c] = f2bf(vq);
    sRelKT[d * 32 + c] = f2bf(vk);
  }
  __syncthreads();

  // ---- Phase 2a: P = q^T x RelQ^T   (64x128, K=32)  -> 4 tiles per wave
  #pragma unroll
  for (int tt = 0; tt < 4; ++tt) {
    int t = wave * 4 + tt;
    int it = t >> 3, dt = t & 7;
    v16bf a  = load_frag_a(qg + it * 16 * 32, 32);
    v16bf bb = load_frag_b(sRelQT + dt * 16 * 32, 32);
    v8f c = {};
    c = __builtin_amdgcn_wmma_f32_16x16x32_bf16(false, a, false, bb, (short)0, c, false, false);
    #pragma unroll
    for (int r = 0; r < 8; ++r)
      sP[(it * 16 + r + 8 * hl) * 132 + dt * 16 + (lane & 15)] = c[r];
  }
  __syncthreads();
  // gather qr[i,j] = P[i, i-j+63]
  #pragma unroll 1
  for (int ii = 0; ii < 16; ++ii) {
    int e = tid + 256 * ii;
    int i = e >> 6, j = e & 63;
    sSim[i * 65 + j] += sc1 * sP[i * 132 + (i - j + 63)] + sh1;
  }
  __syncthreads();

  // ---- Phase 2b: Pk = k^T x RelK^T ; kr[i,j] = Pk[j, j-i+63]
  #pragma unroll
  for (int tt = 0; tt < 4; ++tt) {
    int t = wave * 4 + tt;
    int it = t >> 3, dt = t & 7;
    v16bf a  = load_frag_a(kg + it * 16 * 32, 32);
    v16bf bb = load_frag_b(sRelKT + dt * 16 * 32, 32);
    v8f c = {};
    c = __builtin_amdgcn_wmma_f32_16x16x32_bf16(false, a, false, bb, (short)0, c, false, false);
    #pragma unroll
    for (int r = 0; r < 8; ++r)
      sP[(it * 16 + r + 8 * hl) * 132 + dt * 16 + (lane & 15)] = c[r];
  }
  __syncthreads();
  #pragma unroll 1
  for (int ii = 0; ii < 16; ++ii) {
    int e = tid + 256 * ii;
    int i = e >> 6, j = e & 63;
    sSim[i * 65 + j] += sc2 * sP[j * 132 + (j - i + 63)] + sh2;
  }
  __syncthreads();

  // ---- Phase 3: softmax over j per row i
  if (tid < 64) {
    int i = tid;
    float m = -3.4e38f;
    for (int j = 0; j < 64; ++j) m = fmaxf(m, sSim[i * 65 + j]);
    float s = 0.f;
    for (int j = 0; j < 64; ++j) { float ev = __expf(sSim[i * 65 + j] - m); sSim[i * 65 + j] = ev; s += ev; }
    float inv = 1.f / s;
    for (int j = 0; j < 64; ++j) {
      float p = sSim[i * 65 + j] * inv;
      sSim[i * 65 + j] = p;
      sSimB[i * 72 + j] = f2bf(p);
    }
  }
  __syncthreads();

  // ---- Phase 4: sv[c,i] = sum_j sim[i,j]*v[c,j]  (K=64 -> 2 chained WMMAs)
  #pragma unroll
  for (int tt = 0; tt < 2; ++tt) {
    int t = wave * 2 + tt;
    int it = t >> 2, ct = t & 3;
    v8f c = {};
    #pragma unroll
    for (int kh = 0; kh < 2; ++kh) {
      v16bf a  = load_frag_a(sSimB + it * 16 * 72 + kh * 32, 72);
      v16bf bb = load_frag_b(vg    + ct * 16 * 64 + kh * 32, 64);
      c = __builtin_amdgcn_wmma_f32_16x16x32_bf16(false, a, false, bb, (short)0, c, false, false);
    }
    #pragma unroll
    for (int r = 0; r < 8; ++r) {
      int i   = it * 16 + r + 8 * hl;
      int cch = ct * 16 + (lane & 15);
      sSv[cch * 65 + i] = c[r];
    }
  }

  // ---- Build skewed prob matrix S[i,m] = sim[i, i+63-m] and bf16 rel_v
  #pragma unroll 1
  for (int ii = 0; ii < 32; ++ii) {
    int e = tid + 256 * ii;                       // 8192 entries
    int r0 = e >> 7, m = e & 127;
    int j = r0 + 63 - m;                          // S row r0
    float sv_ = (j >= 0 && j < 64) ? sSim[r0 * 65 + j] : 0.f;
    sST[r0 * 136 + m] = f2bf(sv_);
    float rv = (m < 127) ? relative[(64 + r0) * 127 + m] : 0.f;  // rel_v row r0
    sRelV[r0 * 136 + m] = f2bf(rv);
  }
  __syncthreads();

  // ---- Phase 5: sve = rel_v (64x128) x S^T (128x64) via WMMA, K=128
  #pragma unroll
  for (int tt = 0; tt < 2; ++tt) {
    int t = wave * 2 + tt;
    int ct = t >> 2, it2 = t & 3;
    v8f c = {};
    #pragma unroll
    for (int kh = 0; kh < 4; ++kh) {
      v16bf a  = load_frag_a(sRelV + ct * 16 * 136 + kh * 32, 136);
      v16bf bb = load_frag_b(sST   + it2 * 16 * 136 + kh * 32, 136);
      c = __builtin_amdgcn_wmma_f32_16x16x32_bf16(false, a, false, bb, (short)0, c, false, false);
    }
    #pragma unroll
    for (int r = 0; r < 8; ++r) {
      int cch = ct * 16 + r + 8 * hl;
      int i   = it2 * 16 + (lane & 15);
      sSve[cch * 65 + i] = c[r];
    }
  }
  __syncthreads();

  // ---- Phase 6: out BN on both halves, pair sum, final (L,N,OUT) store
  #pragma unroll 1
  for (int ii = 0; ii < 16; ++ii) {
    int e = tid + 256 * ii;
    int c = e >> 6, i = e & 63;
    int oc = g * 64 + c;
    int o0 = oc * 2, o1 = o0 + 1;
    float a0 = out_g[o0] * rsqrtf(out_v[o0] + EPS);
    float h0 = out_b[o0] - out_m[o0] * a0;
    float a1 = out_g[o1] * rsqrtf(out_v[o1] + EPS);
    float h1 = out_b[o1] - out_m[o1] * a1;
    float val = a0 * sSv[c * 65 + i] + h0 + a1 * sSve[c * 65 + i] + h1;
    int l = w * 64 + i;
    out[((size_t)l * 8 + n) * 512 + oc] = val;
  }
}

// ---------------------------------------------------------------------------
extern "C" void kernel_launch(void* const* d_in, const int* in_sizes, int n_in,
                              void* d_out, int out_size, void* d_ws, size_t ws_size,
                              hipStream_t stream) {
  (void)in_sizes; (void)n_in; (void)out_size; (void)ws_size;
  const float* x        = (const float*)d_in[0];
  const float* qkv_w    = (const float*)d_in[1];
  const float* relative = (const float*)d_in[2];
  const float* qkv_g = (const float*)d_in[3];
  const float* qkv_b = (const float*)d_in[4];
  const float* qkv_m = (const float*)d_in[5];
  const float* qkv_v = (const float*)d_in[6];
  const float* sim_g = (const float*)d_in[7];
  const float* sim_b = (const float*)d_in[8];
  const float* sim_m = (const float*)d_in[9];
  const float* sim_v = (const float*)d_in[10];
  const float* out_g = (const float*)d_in[11];
  const float* out_b = (const float*)d_in[12];
  const float* out_m = (const float*)d_in[13];
  const float* out_v = (const float*)d_in[14];
  float* out = (float*)d_out;

  bf16_t* wq = (bf16_t*)d_ws;                       // 4096 * 64*32 bf16 = 16 MB
  bf16_t* wk = wq + (size_t)4096 * 2048;            // 16 MB
  bf16_t* wv = wk + (size_t)4096 * 2048;            // 32 MB

  dim3 g1(8, 512);
  qkv_gemm_kernel<<<g1, 256, 0, stream>>>(x, qkv_w, qkv_g, qkv_b, qkv_m, qkv_v,
                                          wq, wk, wv);
  attn_kernel<<<4096, 256, 0, stream>>>(wq, wk, wv, relative,
                                        sim_g, sim_b, sim_m, sim_v,
                                        out_g, out_b, out_m, out_v, out);
}
